// ModulatedDCNPack_65249143160856
// MI455X (gfx1250) — matrix operand order repaired and checked by reading, compiled, not run on hardware
//
#include <hip/hip_runtime.h>
#include <hip/hip_bf16.h>

// ---------------------------------------------------------------------------
// DCNv2 (ModulatedDeformConv) for gfx1250:
//   bf16 WMMA implicit-GEMM convs with LDS double-buffering, TDM
//   (tensor_load_to_lds) staging for dense tiles, f32 bilinear sampler.
// ---------------------------------------------------------------------------

typedef __attribute__((ext_vector_type(16))) __bf16 v16bf;
typedef __attribute__((ext_vector_type(8)))  __bf16 v8bf;
typedef __attribute__((ext_vector_type(8)))  float  v8f;

union ABv { v16bf v; v8bf q[2]; };

#define NB     4
#define CIN    256
#define HH     64
#define WW     64
#define PP     4096          // HH*WW
#define KK     9
#define KC     2304          // KK*CIN  (GEMM K)
#define OFFCH  216
#define COUT   256
#define CG     32
#define SSTEP  64            // K elements per LDS stage
#define NSTG   (KC / SSTEP)  // 36 stages

#if defined(__AMDGCN__) && __has_builtin(__builtin_amdgcn_tensor_load_to_lds) && \
    __has_builtin(__builtin_amdgcn_s_wait_tensorcnt)
#define USE_TDM 1
#else
#define USE_TDM 0
#endif

#if USE_TDM
typedef __attribute__((ext_vector_type(4))) unsigned int u32x4;
typedef __attribute__((ext_vector_type(8))) int          i32x8;
typedef __attribute__((ext_vector_type(4))) int          i32x4;

// 2D tile load via Tensor Data Mover. ldsByte: byte offset in the (single)
// LDS allocation; g: element pointer at tile origin; tile0 x tile1 bf16
// elements, row stride strideElems. D# per CDNA5 ISA ch.8.
__device__ __forceinline__ void tdm_load_2d(unsigned ldsByte, const void* g,
                                            unsigned tile0, unsigned tile1,
                                            unsigned strideElems) {
  unsigned long long ga = (unsigned long long)(__UINTPTR_TYPE__)g;
  u32x4 g0;
  g0[0] = 1u;                                              // count=1 (valid)
  g0[1] = ldsByte;                                         // lds_addr
  g0[2] = (unsigned)ga;                                    // global_addr[31:0]
  g0[3] = (unsigned)((ga >> 32) & 0x01ffffffu) | (2u << 30); // [56:32], type=2
  const unsigned td0 = 1u << 20, td1 = 1u << 20;           // generous dims
  i32x8 g1;
  g1[0] = (int)(1u << 16);                                 // data_size = 2B
  g1[1] = (int)((td0 & 0xffffu) << 16);                    // tensor_dim0 lo
  g1[2] = (int)(((td0 >> 16) & 0xffffu) | ((td1 & 0xffffu) << 16));
  g1[3] = (int)(((td1 >> 16) & 0xffffu) | ((tile0 & 0xffffu) << 16));
  g1[4] = (int)(tile1 & 0xffffu);                          // tile_dim1, dim2=0
  g1[5] = (int)strideElems;                                // dim0 stride lo
  g1[6] = 0;
  g1[7] = 0;
  i32x4 z4 = {0, 0, 0, 0};
  i32x8 z8 = {0, 0, 0, 0, 0, 0, 0, 0};
  __builtin_amdgcn_tensor_load_to_lds(g0, g1, z4, z4, z8, 0);
}
#endif

// Synchronous fallback: cooperative global->LDS copy of rows x 64 bf16.
__device__ __forceinline__ void coop_copy(__bf16* ldsDst, const __bf16* g,
                                          int rows, int strideElems, int t) {
  const int chunks = rows * 8;                 // 8 x v8bf per 64-elem row
  for (int c = t; c < chunks; c += 256) {
    const int row = c >> 3, q = c & 7;
    *(v8bf*)(ldsDst + row * SSTEP + q * 8) =
        *(const v8bf*)(g + (__SIZE_TYPE__)row * strideElems + q * 8);
  }
}

// ---------------------------------------------------------------------------
// NCHW f32 -> [b][y][x][c] via 32x32 LDS tile (coalesced both sides)
// ---------------------------------------------------------------------------
__global__ __launch_bounds__(256) void k_tr_f32(
    const float* __restrict__ in, float* __restrict__ out) {
  __shared__ float tile[32][33];
  const int t = threadIdx.x, lane = t & 31, w = t >> 5;
  const int bid = blockIdx.x;               // NB * (CIN/32) * (PP/32) = 4096
  const int pt = bid & 127, ct = (bid >> 7) & 7, b = bid >> 10;
  const int c0 = ct * 32, p0 = pt * 32;
#pragma unroll
  for (int i = 0; i < 4; ++i)
    tile[w + i * 8][lane] =
        in[((__SIZE_TYPE__)(b * CIN + c0 + w + i * 8)) * PP + p0 + lane];
  __syncthreads();
#pragma unroll
  for (int i = 0; i < 4; ++i)
    out[((__SIZE_TYPE__)(b * PP + p0 + w + i * 8)) * CIN + c0 + lane] =
        tile[lane][w + i * 8];
}

__global__ __launch_bounds__(256) void k_tr_bf16(
    const float* __restrict__ in, __bf16* __restrict__ out) {
  __shared__ float tile[32][33];
  const int t = threadIdx.x, lane = t & 31, w = t >> 5;
  const int bid = blockIdx.x;
  const int pt = bid & 127, ct = (bid >> 7) & 7, b = bid >> 10;
  const int c0 = ct * 32, p0 = pt * 32;
#pragma unroll
  for (int i = 0; i < 4; ++i)
    tile[w + i * 8][lane] =
        in[((__SIZE_TYPE__)(b * CIN + c0 + w + i * 8)) * PP + p0 + lane];
  __syncthreads();
#pragma unroll
  for (int i = 0; i < 4; ++i)
    out[((__SIZE_TYPE__)(b * PP + p0 + w + i * 8)) * CIN + c0 + lane] =
        (__bf16)tile[lane][w + i * 8];
}

// [oc][c][ky][kx] f32 -> [oc][k][c] bf16, zero-padded to 256 oc rows
__global__ __launch_bounds__(256) void k_reorder_w(
    const float* __restrict__ w, __bf16* __restrict__ dst, int ocValid) {
  int tid = blockIdx.x * 256 + threadIdx.x;      // 256*KK*CIN threads
  int c   = tid & 255;
  int t   = tid >> 8;                            // oc*9 + k
  int k   = t % KK;
  int oc  = t / KK;
  float v = (oc < ocValid) ? w[(oc * CIN + c) * KK + k] : 0.0f;
  dst[tid] = (__bf16)v;                          // dst[oc*KC + k*256 + c]
}

__global__ __launch_bounds__(256) void k_zero_bf16(__bf16* p, int n) {
  int t = blockIdx.x * 256 + threadIdx.x;
  if (t < n) p[t] = (__bf16)0.0f;
}

// ---------------------------------------------------------------------------
// Offset/mask conv: implicit GEMM, M=216(pad 256), N=16384, K=2304.
// A staged via TDM; B gathered cooperatively (zero-page for OOB taps).
// ---------------------------------------------------------------------------
__global__ __launch_bounds__(256) void k_gemm_off(
    const __bf16* __restrict__ A,       // [256][KC]
    const __bf16* __restrict__ feat,    // [NB][HH][WW][CIN] bf16
    const __bf16* __restrict__ zeroPix, // >=64 zeroed bf16
    const float*  __restrict__ off_b,   // [216]
    float*        __restrict__ om) {    // [NB][216][PP]
  __shared__ __bf16 lds[24576];  // A: 2x4096 elems @0 ; B: 2x8192 elems @8192
  const int t = threadIdx.x, lane = t & 31, wid = t >> 5;
  const int wm = wid & 3, wn = wid >> 2;
  const int r = lane & 15, h = lane >> 4;
  const int mBase = blockIdx.y * 64;
  const int n0b   = blockIdx.x * 128;
  const int b     = n0b >> 12;

  // B-fill geometry: 2 threads per column, 32 elems (64B) each
  const int colT = t >> 1, halfT = t & 1;
  const int pT = (n0b + colT) & 4095;
  const int yT = pT >> 6, xT = pT & 63;

  auto fillB = [&](int stage, int dstElems) {
    const int kk0 = stage * SSTEP;
    const int k   = kk0 >> 8;            // tap index (SSTEP | 256)
    const int c0b = kk0 & 255;           // channel base within tap
    const int ky = k / 3, kx = k - ky * 3;
    const int yy = yT + ky - 1, xx = xT + kx - 1;
    const bool vld = ((unsigned)yy < (unsigned)HH) && ((unsigned)xx < (unsigned)WW);
    const __bf16* src = vld
        ? (feat + ((__SIZE_TYPE__)((b * HH + yy) * WW + xx)) * CIN + c0b)
        : zeroPix;
    __bf16* dst = lds + dstElems + colT * SSTEP + halfT * 32;
    const __bf16* s2 = src + halfT * 32;
#pragma unroll
    for (int q = 0; q < 4; ++q)
      *(v8bf*)(dst + q * 8) = *(const v8bf*)(s2 + q * 8);
  };

  auto fillA = [&](int stage, unsigned ldsByte, int dstElems) {
    const int kk0 = stage * SSTEP;
#if USE_TDM
    (void)dstElems;
    if (wid == 0)
      tdm_load_2d(ldsByte, A + (__SIZE_TYPE__)mBase * KC + kk0, SSTEP, 64u, KC);
#else
    (void)ldsByte;
    coop_copy(lds + dstElems, A + (__SIZE_TYPE__)mBase * KC + kk0, 64, KC, t);
#endif
  };

  fillA(0, 0u, 0);
  fillB(0, 8192);
#if USE_TDM
  if (wid == 0) __builtin_amdgcn_s_wait_tensorcnt(0);
#endif
  __syncthreads();

  v8f acc[4] = {};
  for (int s = 0; s < NSTG; ++s) {
    const int d = s & 1;
    if (s + 1 < NSTG) {
      fillA(s + 1, (unsigned)((1 - d) * 8192), (1 - d) * 4096);
      fillB(s + 1, 8192 + (1 - d) * 8192);
    }
    const __bf16* aT = lds + d * 4096 + (wm * 16 + r) * SSTEP;
    const __bf16* bT = lds + 8192 + d * 8192;
#pragma unroll
    for (int c0 = 0; c0 < SSTEP; c0 += 32) {
      ABv ua;
      ua.q[0] = *(const v8bf*)(aT + c0 + h * 8);
      ua.q[1] = *(const v8bf*)(aT + c0 + 16 + h * 8);
      ABv ub[4];
#pragma unroll
      for (int j = 0; j < 4; ++j) {
        const __bf16* bp = bT + (wn * 64 + j * 16 + r) * SSTEP + c0;
        ub[j].q[0] = *(const v8bf*)(bp + h * 8);
        ub[j].q[1] = *(const v8bf*)(bp + 16 + h * 8);
      }
#pragma unroll
      for (int j = 0; j < 4; ++j)
        acc[j] = __builtin_amdgcn_wmma_f32_16x16x32_bf16(
            false, ua.v, false, ub[j].v, (short)0, acc[j], false, false);
    }
#if USE_TDM
    if (wid == 0) __builtin_amdgcn_s_wait_tensorcnt(0);
#endif
    __syncthreads();
  }

#pragma unroll
  for (int j = 0; j < 4; ++j) {
    int col = n0b + wn * 64 + j * 16 + r;
    int p   = col & 4095;
#pragma unroll
    for (int rr = 0; rr < 8; ++rr) {
      int oc = mBase + wm * 16 + rr + h * 8;
      if (oc < OFFCH)
        om[((__SIZE_TYPE__)b * OFFCH + oc) * PP + p] = acc[j][rr] + off_b[oc];
    }
  }
}

// ---------------------------------------------------------------------------
// Bilinear sampler: wave per (b,dg,k,p), lane = channel-in-group (CG==32).
// ---------------------------------------------------------------------------
__global__ __launch_bounds__(256) void k_sample(
    const float* __restrict__ om,    // [NB][216][PP]
    const float* __restrict__ xh,    // [NB][HH][WW][CIN] f32
    __bf16*      __restrict__ val) { // [NB][PP][KK][CIN] bf16
  int wid  = (blockIdx.x * 256 + threadIdx.x) >> 5;
  int lane = threadIdx.x & 31;
  int dg   = wid & 7;
  int t    = wid >> 3;
  int k    = t % KK;  t /= KK;
  int p    = t & 4095;
  int b    = t >> 12;

  const __SIZE_TYPE__ omb = (__SIZE_TYPE__)b * OFFCH * PP;
  float dy = om[omb + (__SIZE_TYPE__)(dg * 18 + 2 * k)     * PP + p];
  float dx = om[omb + (__SIZE_TYPE__)(dg * 18 + 2 * k + 1) * PP + p];
  float mk = om[omb + (__SIZE_TYPE__)(144 + dg * KK + k)   * PP + p];
  float m  = 1.0f / (1.0f + __expf(-mk));

  int y = p >> 6, x = p & 63;
  int ky = k / 3, kx = k - ky * 3;
  float py = dy + (float)(y + ky - 1);
  float px = dx + (float)(x + kx - 1);
  float y0f = floorf(py), x0f = floorf(px);
  float fy = py - y0f, fx = px - x0f;
  int y0 = (int)y0f, x0 = (int)x0f;

  const int cb = dg * CG + lane;
  auto tap = [&](int yi, int xi) -> float {
    if (((unsigned)yi < (unsigned)HH) && ((unsigned)xi < (unsigned)WW))
      return xh[((__SIZE_TYPE__)((b * HH + yi) * WW + xi)) * CIN + cb];
    return 0.0f;
  };

  float v = (1.0f - fy) * (1.0f - fx) * tap(y0, x0)
          + (1.0f - fy) * fx          * tap(y0, x0 + 1)
          + fy          * (1.0f - fx) * tap(y0 + 1, x0)
          + fy          * fx          * tap(y0 + 1, x0 + 1);

  val[((__SIZE_TYPE__)(b * PP + p) * KK + k) * CIN + cb] = (__bf16)(v * m);
}

// ---------------------------------------------------------------------------
// Main conv: GEMM M=256, N=16384, K=2304. Both operands staged via TDM.
// ---------------------------------------------------------------------------
__global__ __launch_bounds__(256) void k_gemm_main(
    const __bf16* __restrict__ A,      // [256][KC]
    const __bf16* __restrict__ valB,   // [NB*PP][KC]
    const float*  __restrict__ bias,   // [256]
    float*        __restrict__ out) {  // [NB][256][PP]
  __shared__ __bf16 lds[24576];  // A: 2x4096 elems @0 ; B: 2x8192 elems @8192
  const int t = threadIdx.x, lane = t & 31, wid = t >> 5;
  const int wm = wid & 3, wn = wid >> 2;
  const int r = lane & 15, h = lane >> 4;
  const int mBase = blockIdx.y * 64;
  const int n0b   = blockIdx.x * 128;
  const int b     = n0b >> 12;

#if USE_TDM
  if (wid == 0) {
    tdm_load_2d(0u,     A    + (__SIZE_TYPE__)mBase * KC, SSTEP, 64u,  KC);
    tdm_load_2d(16384u, valB + (__SIZE_TYPE__)n0b  * KC,  SSTEP, 128u, KC);
    __builtin_amdgcn_s_wait_tensorcnt(0);
  }
#else
  coop_copy(lds,        A    + (__SIZE_TYPE__)mBase * KC, 64,  KC, t);
  coop_copy(lds + 8192, valB + (__SIZE_TYPE__)n0b  * KC,  128, KC, t);
#endif
  __syncthreads();

  v8f acc[4] = {};
  for (int s = 0; s < NSTG; ++s) {
    const int d = s & 1;
    const int kn = (s + 1) * SSTEP;
    if (s + 1 < NSTG) {
#if USE_TDM
      if (wid == 0) {
        tdm_load_2d((unsigned)((1 - d) * 8192),
                    A + (__SIZE_TYPE__)mBase * KC + kn, SSTEP, 64u, KC);
        tdm_load_2d((unsigned)(16384 + (1 - d) * 16384),
                    valB + (__SIZE_TYPE__)n0b * KC + kn, SSTEP, 128u, KC);
      }
#else
      coop_copy(lds + (1 - d) * 4096,
                A + (__SIZE_TYPE__)mBase * KC + kn, 64, KC, t);
      coop_copy(lds + 8192 + (1 - d) * 8192,
                valB + (__SIZE_TYPE__)n0b * KC + kn, 128, KC, t);
#endif
    }
    const __bf16* aT = lds + d * 4096 + (wm * 16 + r) * SSTEP;
    const __bf16* bT = lds + 8192 + d * 8192;
#pragma unroll
    for (int c0 = 0; c0 < SSTEP; c0 += 32) {
      ABv ua;
      ua.q[0] = *(const v8bf*)(aT + c0 + h * 8);
      ua.q[1] = *(const v8bf*)(aT + c0 + 16 + h * 8);
      ABv ub[4];
#pragma unroll
      for (int j = 0; j < 4; ++j) {
        const __bf16* bp = bT + (wn * 64 + j * 16 + r) * SSTEP + c0;
        ub[j].q[0] = *(const v8bf*)(bp + h * 8);
        ub[j].q[1] = *(const v8bf*)(bp + 16 + h * 8);
      }
#pragma unroll
      for (int j = 0; j < 4; ++j)
        acc[j] = __builtin_amdgcn_wmma_f32_16x16x32_bf16(
            false, ua.v, false, ub[j].v, (short)0, acc[j], false, false);
    }
#if USE_TDM
    if (wid == 0) __builtin_amdgcn_s_wait_tensorcnt(0);
#endif
    __syncthreads();
  }

#pragma unroll
  for (int j = 0; j < 4; ++j) {
    int col = n0b + wn * 64 + j * 16 + r;
    int p   = col & 4095;
#pragma unroll
    for (int rr = 0; rr < 8; ++rr) {
      int oc = mBase + wm * 16 + rr + h * 8;
      out[((__SIZE_TYPE__)b * COUT + oc) * PP + p] = acc[j][rr] + bias[oc];
    }
  }
}

// ---------------------------------------------------------------------------
// Launch
// ---------------------------------------------------------------------------
extern "C" void kernel_launch(void* const* d_in, const int* in_sizes, int n_in,
                              void* d_out, int out_size, void* d_ws, size_t ws_size,
                              hipStream_t stream) {
  (void)in_sizes; (void)n_in; (void)out_size; (void)ws_size;
  const float* x    = (const float*)d_in[0];  // [4,256,64,64]
  const float* ef   = (const float*)d_in[1];  // [4,256,64,64]
  const float* w    = (const float*)d_in[2];  // [256,256,3,3]
  const float* bias = (const float*)d_in[3];  // [256]
  const float* ow   = (const float*)d_in[4];  // [216,256,3,3]
  const float* ob   = (const float*)d_in[5];  // [216]
  float* out = (float*)d_out;

  char* ws = (char*)d_ws;
  float*  xh    = (float*) (ws);                         // 16,777,216 B
  __bf16* fh    = (__bf16*)(ws + 16777216);              //  8,388,608 B
  __bf16* wOff  = (__bf16*)(ws + 25165824);              //  1,179,648 B
  __bf16* wMain = (__bf16*)(ws + 26345472);              //  1,179,648 B
  float*  om    = (float*) (ws + 27525120);              // 14,155,776 B
  __bf16* val   = (__bf16*)(ws + 41680896);              // 75,497,472 B
  __bf16* zpix  = (__bf16*)(ws + 117178368);             //        512 B

  k_tr_f32 <<<4096, 256, 0, stream>>>(x,  xh);
  k_tr_bf16<<<4096, 256, 0, stream>>>(ef, fh);
  k_reorder_w<<<2304, 256, 0, stream>>>(ow, wOff, OFFCH);
  k_reorder_w<<<2304, 256, 0, stream>>>(w,  wMain, COUT);
  k_zero_bf16<<<1, 256, 0, stream>>>(zpix, 256);

  dim3 g1(128, 4);   // N tiles (16384/128) x M blocks (covers 216 rows)
  k_gemm_off<<<g1, 256, 0, stream>>>(wOff, fh, zpix, ob, om);

  k_sample<<<147456, 256, 0, stream>>>(om, xh, val);   // 1,179,648 waves

  dim3 g2(128, 4);
  k_gemm_main<<<g2, 256, 0, stream>>>(wMain, val, bias, out);
}
